// RationaleSelectorModel_41343355191633
// MI455X (gfx1250) — matrix operand order repaired and checked by reading, not compiled, not run on hardware
//
#include <hip/hip_runtime.h>
#include <math.h>

typedef __attribute__((ext_vector_type(16))) __bf16 v16bf;
typedef __attribute__((ext_vector_type(8)))  __bf16 v8bf;
typedef __attribute__((ext_vector_type(8)))  float  v8f;

namespace {
constexpr int   kB   = 16;
constexpr int   kT   = 4096;
constexpr int   kD   = 1024;
constexpr int   kHid  = 1365;
constexpr int   kHidP = 1408;                 // padded to 88 tiles of 16
constexpr int   kWaves = 8;
constexpr int   kTilesPerWave = (kHidP / 16) / kWaves;  // 11
constexpr int   kMRows = 64;                  // rows (tokens) per block
constexpr int   kAst  = kD + 8;               // LDS row stride (bf16 elems), 16B aligned
constexpr float kRho = 0.2f;
constexpr float kLnEps = 1e-5f;
}

__device__ __forceinline__ __bf16 f2bf(float f) {
  unsigned u = __float_as_uint(f);
  unsigned r = u + 0x7FFFu + ((u >> 16) & 1u);   // round to nearest even
  unsigned short s = (unsigned short)(r >> 16);
  __bf16 b; __builtin_memcpy(&b, &s, sizeof(b)); return b;
}

// ---------------- weight prep: fp32 -> padded bf16 ----------------
__global__ void prep_kernel(const float* __restrict__ fc1_w,
                            const float* __restrict__ fc1_b,
                            const float* __restrict__ fc2_w,
                            __bf16* __restrict__ w1,
                            float* __restrict__ b1p,
                            float* __restrict__ w2p,
                            float* __restrict__ ne_out) {
  int idx = blockIdx.x * 256 + threadIdx.x;     // over kHidP * kD
  int h = idx >> 10;
  int d = idx & 1023;
  float v = (h < kHid) ? fc1_w[h * kD + d] : 0.f;
  w1[idx] = f2bf(v);
  if (idx < kHidP) {
    b1p[idx] = (idx < kHid) ? fc1_b[idx] : 0.f;
    w2p[idx] = (idx < kHid) ? fc2_w[idx] : 0.f;
  }
  if (idx == 0) ne_out[0] = 0.f;                // zero entropy accumulator each launch
}

// ---------------- fused mask + LN + fc1(WMMA bf16) + GELU + fc2 ----------------
__global__ __launch_bounds__(256) void fused_scores_kernel(
    const float* __restrict__ emb,  const float* __restrict__ attn,
    const float* __restrict__ ln_w, const float* __restrict__ ln_b,
    const __bf16* __restrict__ w1,  const float* __restrict__ b1p,
    const float* __restrict__ w2p,  const float* __restrict__ fc2_b,
    float* __restrict__ scores) {
  extern __shared__ char smem_raw[];
  __bf16* At = (__bf16*)smem_raw;                               // [kMRows][kAst]
  float* sScore = (float*)(smem_raw + (size_t)kMRows * kAst * sizeof(__bf16));

  const int tid  = threadIdx.x;
  const int w    = tid >> 5;
  const int lane = tid & 31;
  const int r0   = blockIdx.x * kMRows;

  if (tid < kMRows) sScore[tid] = 0.f;

  // ---- Phase 1: masked LayerNorm -> bf16 LDS tile (8 rows per wave) ----
  for (int rr = 0; rr < kMRows / kWaves; ++rr) {
    int lr = w * (kMRows / kWaves) + rr;
    int gr = r0 + lr;
    float attnv = attn[gr];
    const float* ep = emb + (size_t)gr * kD;
    float v[32]; float s = 0.f, sq = 0.f;
    #pragma unroll
    for (int j = 0; j < 32; ++j) {
      float x = ep[lane + j * 32] * attnv;
      v[j] = x; s += x; sq += x * x;
    }
    #pragma unroll
    for (int o = 16; o > 0; o >>= 1) { s += __shfl_xor(s, o, 32); sq += __shfl_xor(sq, o, 32); }
    float mean = s * (1.f / kD);
    float var  = sq * (1.f / kD) - mean * mean;
    float rstd = rsqrtf(var + kLnEps);
    __bf16* arow = At + (size_t)lr * kAst;
    #pragma unroll
    for (int j = 0; j < 32; ++j) {
      int k = lane + j * 32;
      arow[k] = f2bf((v[j] - mean) * rstd * ln_w[k] + ln_b[k]);
    }
  }
  __syncthreads();

  // ---- Phase 2: fc1 via v_wmma_f32_16x16x32_bf16 + GELU + fc2 dot ----
  const int half = lane >> 4;                   // 0: lanes 0-15, 1: lanes 16-31
  const int nl   = lane & 15;
  const __bf16* abase = At + (size_t)nl * kAst + half * 16;

  float part[4][8];
  #pragma unroll
  for (int mt = 0; mt < 4; ++mt)
    #pragma unroll
    for (int vv = 0; vv < 8; ++vv) part[mt][vv] = 0.f;

  for (int t = 0; t < kTilesPerWave; ++t) {
    int nglob = (w * kTilesPerWave + t) * 16 + nl;
    const __bf16* wbase = w1 + (size_t)nglob * kD + half * 16;

    v8f acc[4];
    #pragma unroll
    for (int mt = 0; mt < 4; ++mt)
      #pragma unroll
      for (int vv = 0; vv < 8; ++vv) acc[mt][vv] = 0.f;

    #pragma unroll 2
    for (int kt = 0; kt < 32; ++kt) {
      v16bf bv = *(const v16bf*)(wbase + kt * 32);   // 16 contiguous bf16 (B 32x16 layout)
      #pragma unroll
      for (int mt = 0; mt < 4; ++mt) {
        const __bf16* ap = abase + (size_t)(mt * 16) * kAst + kt * 32;
        v8bf lo = *(const v8bf*)ap;                  // K: k0 + half*16 .. +7
        v8bf hi = *(const v8bf*)(ap + 8);            // K: k0 + half*16 + 8 .. +15
        v16bf af;
        #pragma unroll
        for (int i = 0; i < 8; ++i) { af[i] = lo[i]; af[i + 8] = hi[i]; }
        acc[mt] = __builtin_amdgcn_wmma_f32_16x16x32_bf16(
            false, af, false, bv, (short)0, acc[mt], false, false);
      }
    }

    float b1v = b1p[nglob];
    float w2v = w2p[nglob];
    #pragma unroll
    for (int mt = 0; mt < 4; ++mt)
      #pragma unroll
      for (int vv = 0; vv < 8; ++vv) {
        float x = acc[mt][vv] + b1v;
        float g = 0.5f * x * (1.f + erff(x * 0.70710678118654752f));  // exact GELU
        part[mt][vv] += g * w2v;
      }
  }

  // reduce over the 16 N-lanes of each half-wave; row m = mt*16 + half*8 + vv
  #pragma unroll
  for (int mt = 0; mt < 4; ++mt)
    #pragma unroll
    for (int vv = 0; vv < 8; ++vv) {
      float c = part[mt][vv];
      #pragma unroll
      for (int o = 8; o > 0; o >>= 1) c += __shfl_xor(c, o, 16);
      if (nl == 0) atomicAdd(&sScore[mt * 16 + half * 8 + vv], c);
    }
  __syncthreads();

  if (tid < kMRows) {
    int gr = r0 + tid;
    float sc = sScore[tid] + fc2_b[0];
    if (attn[gr] == 0.f) sc = -1e9f;
    scores[gr] = sc;
  }
}

// ---------------- per-row softmax / entropy / exact top-K ----------------
__device__ __forceinline__ float blockSumF(float v, float* red, int tid) {
  #pragma unroll
  for (int o = 16; o > 0; o >>= 1) v += __shfl_xor(v, o, 32);
  __syncthreads();
  if ((tid & 31) == 0) red[tid >> 5] = v;
  __syncthreads();
  float t = 0.f;
  #pragma unroll
  for (int i = 0; i < 8; ++i) t += red[i];
  return t;
}
__device__ __forceinline__ float blockMaxF(float v, float* red, int tid) {
  #pragma unroll
  for (int o = 16; o > 0; o >>= 1) v = fmaxf(v, __shfl_xor(v, o, 32));
  __syncthreads();
  if ((tid & 31) == 0) red[tid >> 5] = v;
  __syncthreads();
  float t = -3.4e38f;
  #pragma unroll
  for (int i = 0; i < 8; ++i) t = fmaxf(t, red[i]);
  return t;
}
__device__ __forceinline__ int blockSumI(int v, int* red, int tid) {
  #pragma unroll
  for (int o = 16; o > 0; o >>= 1) v += __shfl_xor(v, o, 32);
  __syncthreads();
  if ((tid & 31) == 0) red[tid >> 5] = v;
  __syncthreads();
  int t = 0;
  #pragma unroll
  for (int i = 0; i < 8; ++i) t += red[i];
  return t;
}

__global__ __launch_bounds__(256) void select_kernel(
    const float* __restrict__ scores, const float* __restrict__ attn,
    float* __restrict__ out_g, float* __restrict__ out_z, float* __restrict__ out_ne) {
  __shared__ float redf[8];
  __shared__ int   redi[8];
  __shared__ int   cnts[256];
  const int tid = threadIdx.x;
  const int NE  = kT / 256;                    // 16 contiguous elements per thread
  const size_t base = (size_t)blockIdx.x * kT;

  float s[16]; float cnt = 0.f, mx = -3.4e38f;
  #pragma unroll
  for (int i = 0; i < NE; ++i) {
    int idx = tid * NE + i;
    s[i] = scores[base + idx];
    cnt += attn[base + idx];
    mx = fmaxf(mx, s[i]);
  }
  float Teff = blockSumF(cnt, redf, tid);
  mx = blockMaxF(mx, redf, tid);

  float le = 0.f;
  #pragma unroll
  for (int i = 0; i < NE; ++i) le += __expf(s[i] - mx);
  float S = blockSumF(le, redf, tid);
  float inv = 1.f / S;

  float Kf = fmaxf(1.f, rintf(kRho * Teff));   // round-half-even, clip >= 1
  int Ki = (int)Kf;

  float entl = 0.f;
  float z[16];
  #pragma unroll
  for (int i = 0; i < NE; ++i) {
    float p = __expf(s[i] - mx) * inv;
    entl -= p * __logf(p + 1e-12f);
    z[i] = Kf * p;
  }
  float ENT = blockSumF(entl, redf, tid);

  // order-preserving uint mapping of z
  unsigned u[16];
  #pragma unroll
  for (int i = 0; i < NE; ++i) {
    unsigned ub = __float_as_uint(z[i]);
    u[i] = (ub & 0x80000000u) ? ~ub : (ub | 0x80000000u);
  }

  // MSB radix select of the Ki-th largest value
  unsigned prefix = 0; int remaining = Ki;
  for (int bit = 31; bit >= 0; --bit) {
    unsigned cand = prefix | (1u << bit);
    int c = 0;
    #pragma unroll
    for (int i = 0; i < NE; ++i) c += ((u[i] >> bit) == (cand >> bit)) ? 1 : 0;
    int C = blockSumI(c, redi, tid);
    if (C >= remaining) prefix = cand; else remaining -= C;
  }
  const unsigned theta = prefix;

  int gtl = 0, eql = 0;
  #pragma unroll
  for (int i = 0; i < NE; ++i) { gtl += (u[i] > theta); eql += (u[i] == theta); }
  int GT = blockSumI(gtl, redi, tid);
  __syncthreads();
  cnts[tid] = eql;
  __syncthreads();
  int off = 0;
  for (int j = 0; j < tid; ++j) off += cnts[j];
  const int tiesNeeded = Ki - GT;               // stable: lower index wins ties

  #pragma unroll
  for (int i = 0; i < NE; ++i) {
    int idx = tid * NE + i;
    float g;
    if (u[i] > theta)       g = 1.f;
    else if (u[i] == theta) { g = (off < tiesNeeded) ? 1.f : 0.f; ++off; }
    else                    g = 0.f;
    out_g[base + idx] = g;                      // g = h + (z - sg(z)) == h numerically
    out_z[base + idx] = z[i];
  }
  if (tid == 0)
    atomicAdd(out_ne, ENT / __logf(fmaxf(Teff, 1.f)) * (1.0f / kB));
}

extern "C" void kernel_launch(void* const* d_in, const int* in_sizes, int n_in,
                              void* d_out, int out_size, void* d_ws, size_t ws_size,
                              hipStream_t stream) {
  const float* emb   = (const float*)d_in[0];
  const float* attn  = (const float*)d_in[1];
  const float* ln_w  = (const float*)d_in[2];
  const float* ln_b  = (const float*)d_in[3];
  const float* fc1_w = (const float*)d_in[4];
  const float* fc1_b = (const float*)d_in[5];
  const float* fc2_w = (const float*)d_in[6];
  const float* fc2_b = (const float*)d_in[7];

  float* out = (float*)d_out;
  float* out_g  = out;
  float* out_z  = out + (size_t)kB * kT;
  float* out_ne = out + (size_t)2 * kB * kT;

  char* ws = (char*)d_ws;
  float*  scores = (float*)ws;                                   // 16*4096*4 = 262144 B
  float*  b1p    = (float*)(ws + 262144);                        // 1408*4
  float*  w2p    = (float*)(ws + 262144 + 5632);                 // 1408*4
  __bf16* w1     = (__bf16*)(ws + 262144 + 5632 + 5632);         // 1408*1024*2 (32B aligned)

  prep_kernel<<<(kHidP * kD) / 256, 256, 0, stream>>>(fc1_w, fc1_b, fc2_w, w1, b1p, w2p, out_ne);

  size_t smem = (size_t)kMRows * kAst * sizeof(__bf16) + kMRows * sizeof(float);
  fused_scores_kernel<<<(kB * kT) / kMRows, 256, smem, stream>>>(
      emb, attn, ln_w, ln_b, w1, b1p, w2p, fc2_b, scores);

  select_kernel<<<kB, 256, 0, stream>>>(scores, attn, out_g, out_z, out_ne);
}
